// GAT_57174604644557
// MI455X (gfx1250) — matrix operand order repaired and checked by reading, not compiled, not run on hardware
//
#include <hip/hip_runtime.h>

typedef _Float16 v8h  __attribute__((ext_vector_type(8)));
typedef _Float16 v16h __attribute__((ext_vector_type(16)));
typedef float    v8f  __attribute__((ext_vector_type(8)));

#define DIN    239
#define KIN    256   // DIN padded to multiple of 32
#define DM     128   // model dim
#define NSLOPE_MLP 0.01f
#define NSLOPE_ATT 0.2f

// ---------- ordered-int encoding for float atomicMax ----------
__device__ __forceinline__ unsigned fenc(float f) {
    unsigned u = __float_as_uint(f);
    return (u & 0x80000000u) ? ~u : (u | 0x80000000u);
}
__device__ __forceinline__ float fdec(unsigned u) {
    u = (u & 0x80000000u) ? (u & 0x7FFFFFFFu) : ~u;
    return __uint_as_float(u);
}

// ---------- conversion / init kernels ----------
__global__ __launch_bounds__(256) void cvt_x_kernel(const float* __restrict__ x,
                                                    _Float16* __restrict__ xh,
                                                    int N, int Npad) {
    int i = blockIdx.x * blockDim.x + threadIdx.x;
    if (i >= Npad * KIN) return;
    int n = i >> 8;          // /256
    int k = i & (KIN - 1);
    float v = (n < N && k < DIN) ? x[n * DIN + k] : 0.0f;
    xh[i] = (_Float16)v;
}

// Transpose + convert all three weight matrices to f16 (BT layout: [n][k])
__global__ __launch_bounds__(256) void cvt_w_kernel(const float* __restrict__ W_in,
                                                    const float* __restrict__ W_gat,
                                                    const float* __restrict__ W_h,
                                                    _Float16* __restrict__ WT_in,
                                                    _Float16* __restrict__ WT_gat,
                                                    _Float16* __restrict__ WT_h) {
    int i = blockIdx.x * blockDim.x + threadIdx.x;
    if (i >= DM * KIN) return;
    int n = i >> 8;          // 0..127 output column
    int k = i & (KIN - 1);   // 0..255
    WT_in[n * KIN + k] = (_Float16)((k < DIN) ? W_in[k * DM + n] : 0.0f);
    if (k < DM) {
        WT_gat[n * DM + k] = (_Float16)W_gat[k * DM + n];
        WT_h[n * DM + k]   = (_Float16)W_h[k * DM + n];
    }
}

__global__ __launch_bounds__(256) void init_kernel(float* __restrict__ out_agg,
                                                   unsigned* __restrict__ m_ord,
                                                   float* __restrict__ z,
                                                   int N, int Npad) {
    int i = blockIdx.x * blockDim.x + threadIdx.x;
    if (i < Npad * DM) out_agg[i] = 0.0f;
    if (i < N) { m_ord[i] = 0u; z[i] = 0.0f; }
}

// ---------- WMMA GEMM: D[Npad x 128] = act(A[Npad x K] @ B[K x 128] + bias) ----------
// A: f16 row-major.  BT: f16, B transposed ([128 x K]).  One block = 8 waves =
// one 16-row x 128-col slab; each wave owns a 16x16 tile.  EXEC all-1s.
template <int K, bool ACT, bool OUT16>
__global__ __launch_bounds__(256) void gemm_wmma_kernel(const _Float16* __restrict__ A,
                                                        const _Float16* __restrict__ BT,
                                                        const float* __restrict__ bias,
                                                        float slope,
                                                        float* __restrict__ outf,
                                                        _Float16* __restrict__ outh) {
    const int lane = threadIdx.x & 31;
    const int tn   = threadIdx.x >> 5;          // 0..7 : column tile
    const int tm   = blockIdx.x;                // row tile
    const int l15  = lane & 15;
    const int row  = tm * 16 + l15;             // A row for this lane's fragment
    const int col  = tn * 16 + l15;             // output column
    const int h8   = (lane < 16) ? 0 : 8;       // A-frag K sub-offset (ISA 16-bit A layout)
    const int lh16 = (lane < 16) ? 0 : 8;       // B-frag K sub-offset

    v8f acc = {};
    const _Float16* arow = A + (size_t)row * K;
    const _Float16* brow = BT + (size_t)col * K;

#pragma unroll
    for (int kk = 0; kk < K; kk += 32) {
        // A 16x32 fragment: halves at K = kk+h8..+7 and kk+16+h8..+7 (contiguous 16B each)
        const _Float16* ap = arow + kk + h8;
        v8h a_lo = *(const v8h*)(ap);
        v8h a_hi = *(const v8h*)(ap + 16);
        v16h a = __builtin_shufflevector(a_lo, a_hi, 0, 1, 2, 3, 4, 5, 6, 7,
                                         8, 9, 10, 11, 12, 13, 14, 15);
        // B 32x16 fragment: halves at K = kk+lh16..+7 and kk+16+lh16..+7 (same order as A)
        const _Float16* bp = brow + kk + lh16;
        v8h b_lo = *(const v8h*)(bp);
        v8h b_hi = *(const v8h*)(bp + 16);
        v16h b = __builtin_shufflevector(b_lo, b_hi, 0, 1, 2, 3, 4, 5, 6, 7,
                                         8, 9, 10, 11, 12, 13, 14, 15);
        acc = __builtin_amdgcn_wmma_f32_16x16x32_f16(false, a, false, b,
                                                     (short)0, acc, false, false);
    }

    const float bv = bias ? bias[col] : 0.0f;
#pragma unroll
    for (int r = 0; r < 8; ++r) {
        float v = acc[r] + bv;
        if (ACT) v = (v >= 0.0f) ? v : slope * v;
        size_t o = (size_t)(tm * 16 + r + h8) * DM + col;   // C/D layout: VGPR r -> M=r(+8)
        if (OUT16) outh[o] = (_Float16)v;
        else       outf[o] = v;
    }
}

// ---------- a_src/a_dst: wave-per-node dot products ----------
__global__ __launch_bounds__(256) void att_reduce_kernel(const float* __restrict__ g,
                                                         const float* __restrict__ att_s,
                                                         const float* __restrict__ att_d,
                                                         float* __restrict__ a_s,
                                                         float* __restrict__ a_d, int N) {
    int wv = (blockIdx.x * blockDim.x + threadIdx.x) >> 5;
    int lane = threadIdx.x & 31;
    if (wv >= N) return;
    const float* gr = g + (size_t)wv * DM;
    float s = 0.0f, d = 0.0f;
#pragma unroll
    for (int j = 0; j < 4; ++j) {
        int c = lane + 32 * j;
        float v = gr[c];
        s += v * att_s[c];
        d += v * att_d[c];
    }
#pragma unroll
    for (int off = 16; off > 0; off >>= 1) {
        s += __shfl_down(s, off);
        d += __shfl_down(d, off);
    }
    if (lane == 0) { a_s[wv] = s; a_d[wv] = d; }
}

// ---------- edge pass 1: logits + segment max (ordered-int atomicMax) ----------
__global__ __launch_bounds__(256) void edge_logit_kernel(const int* __restrict__ ei,
                                                         const float* __restrict__ a_s,
                                                         const float* __restrict__ a_d,
                                                         float* __restrict__ e,
                                                         unsigned* __restrict__ m_ord,
                                                         int E, int N) {
    int t = blockIdx.x * blockDim.x + threadIdx.x;
    if (t >= E + N) return;
    int s, d;
    if (t < E) { s = ei[t]; d = ei[E + t]; } else { s = d = t - E; }
    float v = a_s[s] + a_d[d];
    v = (v >= 0.0f) ? v : NSLOPE_ATT * v;
    e[t] = v;
    atomicMax(&m_ord[d], fenc(v));
}

// ---------- edge pass 2: w = exp(e - m[dst]); z[dst] += w ----------
__global__ __launch_bounds__(256) void edge_exp_kernel(const int* __restrict__ ei,
                                                       float* __restrict__ e,
                                                       const unsigned* __restrict__ m_ord,
                                                       float* __restrict__ z,
                                                       int E, int N) {
    int t = blockIdx.x * blockDim.x + threadIdx.x;
    if (t >= E + N) return;
    int d = (t < E) ? ei[E + t] : (t - E);
    float w = __expf(e[t] - fdec(m_ord[d]));
    e[t] = w;
    atomicAdd(&z[d], w);
}

// ---------- edge pass 3: wave-per-edge weighted scatter of g[src] ----------
__global__ __launch_bounds__(256) void edge_scatter_kernel(const int* __restrict__ ei,
                                                           const float* __restrict__ w,
                                                           const float* __restrict__ z,
                                                           const float* __restrict__ g,
                                                           float* __restrict__ out_agg,
                                                           int E, int N) {
    int gid = blockIdx.x * blockDim.x + threadIdx.x;
    int wv = gid >> 5, lane = gid & 31;
    if (wv >= E + N) return;
    int s, d;
    if (wv < E) { s = ei[wv]; d = ei[E + wv]; } else { s = d = wv - E; }
    float alpha = w[wv] / z[d];
    const float* gr = g + (size_t)s * DM;
    float* orow = out_agg + (size_t)d * DM;
#pragma unroll
    for (int j = 0; j < 4; ++j) {
        int c = lane + 32 * j;
        atomicAdd(&orow[c], alpha * gr[c]);
    }
}

// ---------- (out_agg + b_gat) -> f16 for the next GEMM ----------
__global__ __launch_bounds__(256) void cvt_out_kernel(const float* __restrict__ out_agg,
                                                      const float* __restrict__ b_gat,
                                                      _Float16* __restrict__ outh, int total) {
    int i = blockIdx.x * blockDim.x + threadIdx.x;
    if (i >= total) return;
    outh[i] = (_Float16)(out_agg[i] + b_gat[i & (DM - 1)]);
}

// ---------- final 128 -> 2 projection: wave-per-node ----------
__global__ __launch_bounds__(256) void final_proj_kernel(const float* __restrict__ h2,
                                                         const float* __restrict__ W,
                                                         const float* __restrict__ b,
                                                         float* __restrict__ out, int N) {
    int wv = (blockIdx.x * blockDim.x + threadIdx.x) >> 5;
    int lane = threadIdx.x & 31;
    if (wv >= N) return;
    const float* hr = h2 + (size_t)wv * DM;
    float a0 = 0.0f, a1 = 0.0f;
#pragma unroll
    for (int j = 0; j < 4; ++j) {
        int c = lane + 32 * j;
        float v = hr[c];
        a0 += v * W[c * 2 + 0];
        a1 += v * W[c * 2 + 1];
    }
#pragma unroll
    for (int off = 16; off > 0; off >>= 1) {
        a0 += __shfl_down(a0, off);
        a1 += __shfl_down(a1, off);
    }
    if (lane == 0) {
        out[wv * 2 + 0] = a0 + b[0];
        out[wv * 2 + 1] = a1 + b[1];
    }
}

extern "C" void kernel_launch(void* const* d_in, const int* in_sizes, int n_in,
                              void* d_out, int out_size, void* d_ws, size_t ws_size,
                              hipStream_t stream) {
    const float* x        = (const float*)d_in[0];
    const int*   ei       = (const int*)d_in[1];
    /* d_in[2] edge_type: unused by reference */
    const float* W_in     = (const float*)d_in[3];
    const float* b_in     = (const float*)d_in[4];
    const float* W_gat    = (const float*)d_in[5];
    const float* att_src  = (const float*)d_in[6];
    const float* att_dst  = (const float*)d_in[7];
    const float* b_gat    = (const float*)d_in[8];
    const float* W_h      = (const float*)d_in[9];
    const float* b_h      = (const float*)d_in[10];
    const float* W_out    = (const float*)d_in[11];
    const float* b_out    = (const float*)d_in[12];
    float*       out      = (float*)d_out;

    const int N    = in_sizes[0] / DIN;   // 50000
    const int E    = in_sizes[1] / 2;     // 600000
    const int T    = E + N;               // edges incl. self loops
    const int mt   = (N + 15) / 16;
    const int Npad = mt * 16;             // 50000 is already 3125*16

    // ---- workspace carve-up (256B aligned), with buffer reuse ----
    char* ws = (char*)d_ws;
    size_t off = 0;
    auto take = [&](size_t bytes) {
        void* p = ws + off;
        off += (bytes + 255) & ~(size_t)255;
        return p;
    };
    // buffer A: xh (Npad*256 f16) later reused as out_agg (Npad*128 f32) -- same size
    _Float16* xh      = (_Float16*)take((size_t)Npad * KIN * sizeof(_Float16));
    float*    out_agg = (float*)xh;
    // buffer B: h16 (Npad*128 f16) later reused as outh
    _Float16* h16     = (_Float16*)take((size_t)Npad * DM * sizeof(_Float16));
    _Float16* outh    = h16;
    // buffer C: g (Npad*128 f32) later reused as h2
    float*    g       = (float*)take((size_t)Npad * DM * sizeof(float));
    float*    h2      = g;
    _Float16* WT_in   = (_Float16*)take((size_t)DM * KIN * sizeof(_Float16));
    _Float16* WT_gat  = (_Float16*)take((size_t)DM * DM * sizeof(_Float16));
    _Float16* WT_h    = (_Float16*)take((size_t)DM * DM * sizeof(_Float16));
    float*    a_s     = (float*)take((size_t)N * sizeof(float));
    float*    a_d     = (float*)take((size_t)N * sizeof(float));
    unsigned* m_ord   = (unsigned*)take((size_t)N * sizeof(unsigned));
    float*    z       = (float*)take((size_t)N * sizeof(float));
    float*    ew      = (float*)take((size_t)T * sizeof(float));

    const int B256 = 256;
    auto blk = [&](long long total) { return (int)((total + B256 - 1) / B256); };

    // 1) conversions
    cvt_x_kernel<<<blk((long long)Npad * KIN), B256, 0, stream>>>(x, xh, N, Npad);
    cvt_w_kernel<<<blk((long long)DM * KIN), B256, 0, stream>>>(W_in, W_gat, W_h,
                                                                WT_in, WT_gat, WT_h);
    // 2) h = leaky(x @ W_in + b_in)  [f16 out]
    gemm_wmma_kernel<KIN, true, true><<<mt, B256, 0, stream>>>(xh, WT_in, b_in,
                                                               NSLOPE_MLP, nullptr, h16);
    // 3) g = h @ W_gat  [f32 out]
    gemm_wmma_kernel<DM, false, false><<<mt, B256, 0, stream>>>(h16, WT_gat, nullptr,
                                                                0.0f, g, nullptr);
    // 4) a_src / a_dst
    att_reduce_kernel<<<blk((long long)N * 32), B256, 0, stream>>>(g, att_src, att_dst,
                                                                   a_s, a_d, N);
    // 5) init accumulators (out_agg reuses xh, which is no longer needed)
    init_kernel<<<blk((long long)Npad * DM), B256, 0, stream>>>(out_agg, m_ord, z, N, Npad);
    // 6) edge softmax + scatter
    edge_logit_kernel<<<blk(T), B256, 0, stream>>>(ei, a_s, a_d, ew, m_ord, E, N);
    edge_exp_kernel<<<blk(T), B256, 0, stream>>>(ei, ew, m_ord, z, E, N);
    edge_scatter_kernel<<<blk((long long)T * 32), B256, 0, stream>>>(ei, ew, z, g,
                                                                     out_agg, E, N);
    // 7) (out_agg + b_gat) -> f16
    cvt_out_kernel<<<blk((long long)Npad * DM), B256, 0, stream>>>(out_agg, b_gat, outh,
                                                                   Npad * DM);
    // 8) h2 = leaky(outh @ W_h + b_h)  [f32 out, reuses g's buffer]
    gemm_wmma_kernel<DM, true, false><<<mt, B256, 0, stream>>>(outh, WT_h, b_h,
                                                               NSLOPE_MLP, h2, nullptr);
    // 9) final projection 128 -> 2
    final_proj_kernel<<<blk((long long)N * 32), B256, 0, stream>>>(h2, W_out, b_out, out, N);
}
